// MultiHeadAttention_86182813762230
// MI455X (gfx1250) — matrix operand order repaired and checked by reading, not compiled, not run on hardware
//
#include <hip/hip_runtime.h>

// ---------------------------------------------------------------------------
// MI455X (gfx1250) MHA forward: bf16 WMMA GEMMs + TDM-staged f32 softmax.
// out  = (softmax(QWq^T (KWk^T)^T / sqrt(dk)) (VWv^T)) Wo^T + b
// d_out = [ out : B*S*D f32 ][ attn : B*H*S*S f32 ]
// ---------------------------------------------------------------------------

typedef __attribute__((ext_vector_type(16))) __bf16 v16bf;
typedef __attribute__((ext_vector_type(8)))  __bf16 v8bf;
typedef __attribute__((ext_vector_type(4)))  __bf16 v4bf;
typedef __attribute__((ext_vector_type(8)))  float  v8f;
typedef __attribute__((ext_vector_type(4)))  float  v4f;
typedef __attribute__((ext_vector_type(4)))  unsigned int u32x4;
typedef __attribute__((ext_vector_type(8)))  unsigned int u32x8;

constexpr int B_  = 2;
constexpr int S_  = 2048;
constexpr int D_  = 1024;
constexpr int H_  = 8;
constexpr int DK_ = 128;

static __device__ inline __bf16 f2bf(float f) {
    union { float f; unsigned u; } x; x.f = f;
    unsigned r = x.u + 0x7FFFu + ((x.u >> 16) & 1u);   // round-to-nearest-even
    unsigned short h = (unsigned short)(r >> 16);
    return __builtin_bit_cast(__bf16, h);
}

union ABf { v16bf v; v8bf h[2]; };

// ---------------------------------------------------------------------------
// f32 -> bf16 bulk convert, 4 elements/thread
// ---------------------------------------------------------------------------
__global__ __launch_bounds__(256) void cvt_f32_bf16(const float* __restrict__ src,
                                                    __bf16* __restrict__ dst,
                                                    size_t n4) {
    size_t i = (size_t)blockIdx.x * blockDim.x + threadIdx.x;
    if (i < n4) {
        v4f f = ((const v4f*)src)[i];
        v4bf o;
#pragma unroll
        for (int e = 0; e < 4; ++e) o[e] = f2bf(f[e]);
        ((v4bf*)dst)[i] = o;
    }
}

// ---------------------------------------------------------------------------
// WMMA GEMM: C = A[M,K] @ Bm[N,K]^T (torch Linear layout), batched over
// blockIdx.z (element strides strideA/strideB/strideO).
// Wave tile 32x64 (2x4 WMMA f32 accs), 8 waves/block -> 128x128 block tile.
// OUT_MODE: 0 = bf16 head-split [B,H,S,dk] (+bias)            (q,k proj)
//           4 = bf16 head-split transposed [B,H,dk,S] (+bias) (v proj)
//           2 = f32 row-major * scale, batched                (scores)
//           3 = bf16 concat layout [B,S,H*dk]                 (context)
//           1 = f32 row-major + bias                          (out proj)
// A_IS_F32: A is f32, converted to bf16 in-register (ctx GEMM reads attn).
// ---------------------------------------------------------------------------
template <int OUT_MODE, bool A_IS_F32, int N, int K>
__global__ __launch_bounds__(256)
void gemm_wmma_bf16(const void* __restrict__ Av, const __bf16* __restrict__ Bm,
                    const float* __restrict__ bias, void* __restrict__ Ov,
                    long long strideA, long long strideB, long long strideO,
                    float scale)
{
    const int lane = threadIdx.x & 31;
    const int wave = threadIdx.x >> 5;
    const int z    = blockIdx.z;
    const int wm   = blockIdx.x * 128 + (wave >> 1) * 32;
    const int wn   = blockIdx.y * 128 + (wave & 1)  * 64;

    const int  mrow = lane & 15;
    const bool hi   = lane >= 16;

    const __bf16* Ab = (const __bf16*)Av + (size_t)z * strideA;
    const float*  Af = (const float* )Av + (size_t)z * strideA;
    const __bf16* Bp = Bm + (size_t)z * strideB;

    v8f acc[2][4] = {};

    const int akb_off = hi ? 8  : 0;   // A frag: K runs [kb..kb+7],[kb+16..kb+23]
    const int bkb_off = hi ? 8  : 0;   // B frag: K runs [kb..kb+7],[kb+16..kb+23]

    for (int k = 0; k < K; k += 32) {
        const int akb = k + akb_off;
        const int bkb = k + bkb_off;

        v16bf afrag[2];
#pragma unroll
        for (int ti = 0; ti < 2; ++ti) {
            const int m = wm + ti * 16 + mrow;
            if (A_IS_F32) {
                const float* pa = Af + (size_t)m * K + akb;
                v8f f0 = *(const v8f*)(pa);
                v8f f1 = *(const v8f*)(pa + 16);
                ABf t;
#pragma unroll
                for (int e = 0; e < 8; ++e) {
                    t.v[e]     = f2bf(f0[e]);
                    t.v[8 + e] = f2bf(f1[e]);
                }
                afrag[ti] = t.v;
            } else {
                const __bf16* pa = Ab + (size_t)m * K + akb;
                ABf t;
                t.h[0] = *(const v8bf*)(pa);
                t.h[1] = *(const v8bf*)(pa + 16);
                afrag[ti] = t.v;
            }
        }

        v16bf bfrag[4];
#pragma unroll
        for (int tj = 0; tj < 4; ++tj) {
            const int n = wn + tj * 16 + mrow;
            ABf tb;
            tb.h[0] = *(const v8bf*)(Bp + (size_t)n * K + bkb);
            tb.h[1] = *(const v8bf*)(Bp + (size_t)n * K + bkb + 16);
            bfrag[tj] = tb.v;
        }

        // speculative prefetch of next-next B K-chunk (drop-on-fault)
        __builtin_prefetch((const void*)(Bp + (size_t)(wn + mrow) * K + k + 64), 0, 1);

#pragma unroll
        for (int ti = 0; ti < 2; ++ti)
#pragma unroll
            for (int tj = 0; tj < 4; ++tj)
                acc[ti][tj] = __builtin_amdgcn_wmma_f32_16x16x32_bf16(
                    false, afrag[ti], false, bfrag[tj],
                    (short)0, acc[ti][tj], false, false);
    }

    const int mstore = hi ? 8 : 0;
#pragma unroll
    for (int tj = 0; tj < 4; ++tj) {
        const int n = wn + tj * 16 + mrow;
        const float bn = (OUT_MODE == 0 || OUT_MODE == 4 || OUT_MODE == 1) ? bias[n] : 0.0f;
#pragma unroll
        for (int ti = 0; ti < 2; ++ti) {
#pragma unroll
            for (int r = 0; r < 8; ++r) {
                const int m = wm + ti * 16 + mstore + r;
                float val = acc[ti][tj][r];
                if (OUT_MODE == 0 || OUT_MODE == 4) {
                    val += bn;
                    const int b = m / S_, s = m % S_, h = n / DK_, d = n % DK_;
                    __bf16* Ob = (__bf16*)Ov;
                    const size_t off = (OUT_MODE == 0)
                        ? ((((size_t)b * H_ + h) * S_ + s) * DK_ + d)   // [B,H,S,dk]
                        : ((((size_t)b * H_ + h) * DK_ + d) * S_ + s);  // [B,H,dk,S]
                    Ob[off] = f2bf(val);
                } else if (OUT_MODE == 1) {
                    float* Of = (float*)Ov;
                    Of[(size_t)m * N + n] = val + bn;
                } else if (OUT_MODE == 2) {
                    float* Of = (float*)Ov + (size_t)z * strideO;
                    Of[(size_t)m * N + n] = val * scale;
                } else { // 3: context -> [B, S, H*dk]
                    const int b = z / H_, h = z % H_;
                    __bf16* Ob = (__bf16*)Ov;
                    const size_t off =
                        ((size_t)b * S_ + m) * (size_t)D_ + (size_t)h * DK_ + n;
                    Ob[off] = f2bf(val);
                }
            }
        }
    }
}

// ---------------------------------------------------------------------------
// In-place row softmax, one block per 2048-float row.  The row is staged into
// LDS once by the Tensor Data Mover (D# built per CDNA5 ISA ch.8), then the
// max / exp+sum passes read LDS instead of re-reading global.
// ---------------------------------------------------------------------------
__global__ __launch_bounds__(256) void softmax_rows_tdm(float* __restrict__ p) {
    constexpr int L = S_;
    __shared__ float rowbuf[L];     // 8 KB, written by TDM
    __shared__ float red[256];

    float* grow = p + (size_t)blockIdx.x * L;
    const int t = threadIdx.x;

    if (t == 0) {
        const unsigned long long ga = (unsigned long long)(const void*)grow;
        const unsigned lds_off = (unsigned)(size_t)(void*)&rowbuf[0];
        u32x4 g0;
        g0[0] = 1u;                                        // count=1 (valid user D#)
        g0[1] = lds_off;                                   // lds_addr
        g0[2] = (unsigned)(ga & 0xFFFFFFFFu);              // global_addr[31:0]
        g0[3] = (unsigned)((ga >> 32) & 0x01FFFFFFu)       // global_addr[56:32]
              | 0x80000000u;                               // type=2 ("image")
        u32x8 g1;
        g1[0] = 2u << 16;                                  // data_size=4B, wg_mask=0
        g1[1] = ((unsigned)L & 0xFFFFu) << 16;             // tensor_dim0[15:0]
        g1[2] = ((unsigned)L >> 16) | (1u << 16);          // tensor_dim0[31:16], tensor_dim1=1
        g1[3] = ((unsigned)L) << 16;                       // tile_dim0 = L
        g1[4] = 1u;                                        // tile_dim1=1, tile_dim2=0
        g1[5] = (unsigned)L;                               // tensor_dim0_stride[31:0]
        g1[6] = 0u;
        g1[7] = 0u;
        asm volatile("tensor_load_to_lds %0, %1" :: "s"(g0), "s"(g1) : "memory");
    }
    __builtin_amdgcn_s_wait_tensorcnt(0);
    __syncthreads();

    float mx = -3.402823466e38f;
    for (int i = t; i < L; i += 256) mx = fmaxf(mx, rowbuf[i]);
    red[t] = mx;
    __syncthreads();
    for (int s = 128; s > 0; s >>= 1) {
        if (t < s) red[t] = fmaxf(red[t], red[t + s]);
        __syncthreads();
    }
    mx = red[0];
    __syncthreads();

    float sum = 0.f;
    for (int i = t; i < L; i += 256) {
        const float e = __expf(rowbuf[i] - mx);
        rowbuf[i] = e;
        sum += e;
    }
    red[t] = sum;
    __syncthreads();
    for (int s = 128; s > 0; s >>= 1) {
        if (t < s) red[t] += red[t + s];
        __syncthreads();
    }
    const float inv = 1.0f / red[0];
    for (int i = t; i < L; i += 256) grow[i] = rowbuf[i] * inv;
}

// ---------------------------------------------------------------------------
extern "C" void kernel_launch(void* const* d_in, const int* in_sizes, int n_in,
                              void* d_out, int out_size, void* d_ws, size_t ws_size,
                              hipStream_t stream) {
    const float* Q   = (const float*)d_in[0];
    const float* Kin = (const float*)d_in[1];
    const float* V   = (const float*)d_in[2];
    const float* Wq  = (const float*)d_in[3];
    const float* bq  = (const float*)d_in[4];
    const float* Wk  = (const float*)d_in[5];
    const float* bk  = (const float*)d_in[6];
    const float* Wv  = (const float*)d_in[7];
    const float* bv  = (const float*)d_in[8];
    const float* Wo  = (const float*)d_in[9];
    const float* bo  = (const float*)d_in[10];

    const int M = B_ * S_;                 // 4096
    const size_t MD = (size_t)M * D_;      // 4,194,304
    const size_t DD = (size_t)D_ * D_;     // 1,048,576

    // workspace layout (bf16 elements)
    __bf16* ws  = (__bf16*)d_ws;
    __bf16* Qb  = ws;                 // [B*S, D]
    __bf16* Kb  = Qb  + MD;
    __bf16* Vb  = Kb  + MD;
    __bf16* Wqb = Vb  + MD;           // [D, D] each
    __bf16* Wkb = Wqb + DD;
    __bf16* Wvb = Wkb + DD;
    __bf16* Wob = Wvb + DD;
    __bf16* qh  = Wob + DD;           // [B,H,S,dk]
    __bf16* kh  = qh  + MD;           // [B,H,S,dk]
    __bf16* vT  = kh  + MD;           // [B,H,dk,S]
    __bf16* ctx = vT  + MD;           // [B,S,D]

    float* out  = (float*)d_out;      // [B,S,D]
    float* attn = out + MD;           // [B,H,S,S]

    const dim3 blk(256);

    // f32 -> bf16 converts (4 elems/thread)
    cvt_f32_bf16<<<dim3((unsigned)(MD / 4 / 256)), blk, 0, stream>>>(Q,   Qb,  MD / 4);
    cvt_f32_bf16<<<dim3((unsigned)(MD / 4 / 256)), blk, 0, stream>>>(Kin, Kb,  MD / 4);
    cvt_f32_bf16<<<dim3((unsigned)(MD / 4 / 256)), blk, 0, stream>>>(V,   Vb,  MD / 4);
    cvt_f32_bf16<<<dim3((unsigned)(DD / 4 / 256)), blk, 0, stream>>>(Wq,  Wqb, DD / 4);
    cvt_f32_bf16<<<dim3((unsigned)(DD / 4 / 256)), blk, 0, stream>>>(Wk,  Wkb, DD / 4);
    cvt_f32_bf16<<<dim3((unsigned)(DD / 4 / 256)), blk, 0, stream>>>(Wv,  Wvb, DD / 4);
    cvt_f32_bf16<<<dim3((unsigned)(DD / 4 / 256)), blk, 0, stream>>>(Wo,  Wob, DD / 4);

    // projections: q,k head-split; v head-split transposed
    gemm_wmma_bf16<0, false, D_, D_><<<dim3(M / 128, D_ / 128, 1), blk, 0, stream>>>(
        Qb, Wqb, bq, qh, 0, 0, 0, 1.0f);
    gemm_wmma_bf16<0, false, D_, D_><<<dim3(M / 128, D_ / 128, 1), blk, 0, stream>>>(
        Kb, Wkb, bk, kh, 0, 0, 0, 1.0f);
    gemm_wmma_bf16<4, false, D_, D_><<<dim3(M / 128, D_ / 128, 1), blk, 0, stream>>>(
        Vb, Wvb, bv, vT, 0, 0, 0, 1.0f);

    // scores = q @ k^T / sqrt(dk), batched over B*H, straight into attn output
    const float inv_sqrt_dk = 0.0883883476483184f;  // 1/sqrt(128)
    gemm_wmma_bf16<2, false, S_, DK_><<<dim3(S_ / 128, S_ / 128, B_ * H_), blk, 0, stream>>>(
        qh, kh, nullptr, attn,
        (long long)S_ * DK_, (long long)S_ * DK_, (long long)S_ * S_,
        inv_sqrt_dk);

    // softmax rows in place (TDM stages each row into LDS)
    softmax_rows_tdm<<<dim3(B_ * H_ * S_), blk, 0, stream>>>(attn);

    // ctx = attn @ v  (A read as f32, converted in-register; B = v^T rows)
    gemm_wmma_bf16<3, true, DK_, S_><<<dim3(S_ / 128, DK_ / 128, B_ * H_), blk, 0, stream>>>(
        attn, vT, nullptr, ctx,
        (long long)S_ * S_, (long long)DK_ * S_, 0,
        1.0f);

    // out = ctx @ Wo^T + bo
    gemm_wmma_bf16<1, false, D_, D_><<<dim3(M / 128, D_ / 128, 1), blk, 0, stream>>>(
        ctx, Wob, bo, out, 0, 0, 0, 1.0f);
}
